// GATv2_model_54606214201744
// MI455X (gfx1250) — matrix operand order repaired and checked by reading, not compiled, not run on hardware
//
#include <hip/hip_runtime.h>
#include <hip/hip_bf16.h>
#include <hip/hip_fp16.h>

#define N_NODES 51200
#define N_EDGES 1638400
#define N_GRAPH 512
#define DIM_IN  16
#define HEADS   4
#define CHAN    32
#define HCDIM   128
#define SLOPE   0.2f
#define MTILES  4            // 16-row tiles per block (64 rows/block)
// 1/sqrt(1 + 1e-5)
#define INV_SQRT_1PEPS 0.99999500003749975f
#define ENC_NEG_INF 0x007FFFFFu

typedef __attribute__((ext_vector_type(16))) _Float16 v16h;
typedef __attribute__((ext_vector_type(8)))  float    v8f;

// ---- monotonic float<->uint encoding for atomicMax on floats ----
__device__ __forceinline__ unsigned fenc(float f) {
  unsigned u = __float_as_uint(f);
  return (u & 0x80000000u) ? ~u : (u | 0x80000000u);
}
__device__ __forceinline__ float fdec(unsigned e) {
  return (e & 0x80000000u) ? __uint_as_float(e & 0x7fffffffu)
                           : __uint_as_float(~e);
}

// =====================================================================
// WMMA GEMM: out[M x 128] = epilogue(A[M x K] @ W[K x 128] + bias)
//   MODE 0: out = acc + bias
//   MODE 1: out = relu((acc+bias) * gamma[c]*INV_SQRT_1PEPS + beta[c])
// K, MODE are compile-time: K-loop fully unrolls, bounds guards constant-
// fold (K=128: none at all; K=16: one cndmask set on the B fragment).
// Block = 256 threads = 8 waves; wave w owns column tile [16w, 16w+16).
// Block owns 64 rows (4 M-tiles); B fragment reused across the 4 tiles.
// EXEC stays full everywhere -> WMMA legal.
// Fragment layouts per CDNA5 ISA 7.12.2 (wave32).
// =====================================================================
template <int K, int MODE>
__global__ __launch_bounds__(256)
void gemm_n128_wmma(const float* __restrict__ A,
                    const float* __restrict__ W128,
                    const float* __restrict__ bias,
                    const float* __restrict__ gamma,
                    const float* __restrict__ beta,
                    float* __restrict__ out)
{
  const int lane = threadIdx.x & 31;
  const int wave = threadIdx.x >> 5;       // 0..7 -> column tile
  const int m0   = blockIdx.x * (16 * MTILES);
  const int n    = lane & 15;
  const int ncol = wave * 16 + n;          // output column
  const int hi   = lane >> 4;              // 0 or 1 (lane group)
  const int kbA  = hi * 8;                 // A: lanes 16-31 hold K+8 group
  const int kbB  = hi * 8;                 // B: lanes 16-31 hold K+8 group

  v8f acc[MTILES];
  {
    v8f z = {};
    #pragma unroll
    for (int mt = 0; mt < MTILES; ++mt) acc[mt] = z;
  }

  #pragma unroll
  for (int k0 = 0; k0 < K; k0 += 32) {
    const bool run2 = (K - k0) >= 32;      // compile-time constant

    // ---- B fragment: elem e<8 -> K = k0 + kbB + e, e>=8 -> K = k0 + 16 + kbB + (e-8), col = ncol.
    // run 1 valid iff (k0+kbB) < K; run 2 valid iff run2.
    v16h b;
    {
      const bool bvalid = (k0 + kbB) < K;  // folds away for K=128
      const int  kb     = bvalid ? (k0 + kbB) : 0;
      const float* wp   = W128 + (size_t)kb * HCDIM + ncol;
      #pragma unroll
      for (int e = 0; e < 16; ++e) {
        float v = (e < 8 || run2) ? wp[(size_t)(e < 8 ? e : e + 8) * HCDIM] : 0.0f;
        b[e] = (_Float16)(bvalid ? v : 0.0f);
      }
    }

    // ---- 4 M-tiles share this B fragment ----
    #pragma unroll
    for (int mt = 0; mt < MTILES; ++mt) {
      const float* arow = A + (size_t)(m0 + mt * 16 + n) * K;
      v16h a;
      {   // run 1: K = k0 + kbA + [0..7]  (always in-bounds)
        const float4* p = (const float4*)(arow + k0 + kbA);
        float4 q0 = p[0], q1 = p[1];
        a[0]=(_Float16)q0.x; a[1]=(_Float16)q0.y; a[2]=(_Float16)q0.z; a[3]=(_Float16)q0.w;
        a[4]=(_Float16)q1.x; a[5]=(_Float16)q1.y; a[6]=(_Float16)q1.z; a[7]=(_Float16)q1.w;
      }
      if (run2) {  // run 2: K = k0 + 16 + kbA + [0..7]
        const float4* p = (const float4*)(arow + k0 + 16 + kbA);
        float4 q0 = p[0], q1 = p[1];
        a[8] =(_Float16)q0.x; a[9] =(_Float16)q0.y; a[10]=(_Float16)q0.z; a[11]=(_Float16)q0.w;
        a[12]=(_Float16)q1.x; a[13]=(_Float16)q1.y; a[14]=(_Float16)q1.z; a[15]=(_Float16)q1.w;
      } else {
        #pragma unroll
        for (int e = 8; e < 16; ++e) a[e] = (_Float16)0.0f;
      }
      acc[mt] = __builtin_amdgcn_wmma_f32_16x16x32_f16(
          /*neg_a=*/false, a, /*neg_b=*/false, b,
          /*c_mod=*/(short)0, acc[mt], /*reuse_a=*/false, /*reuse_b=*/false);
    }
  }

  // C/D layout: VGPR r -> M = hi*8 + r, N = lane%16
  float bv = bias[ncol];
  float gm = 0.0f, bt = 0.0f;
  if (MODE == 1) { gm = gamma[ncol] * INV_SQRT_1PEPS; bt = beta[ncol]; }
  #pragma unroll
  for (int mt = 0; mt < MTILES; ++mt) {
    #pragma unroll
    for (int r = 0; r < 8; ++r) {
      int row = m0 + mt * 16 + hi * 8 + r;
      float v = acc[mt][r] + bv;
      if (MODE == 1) v = fmaxf(fmaf(v, gm, bt), 0.0f);
      out[(size_t)row * HCDIM + ncol] = v;
    }
  }
}

// =====================================================================
// Edge-phase kernels: one thread per (edge, head).  t = e*4 + h.
// =====================================================================
__global__ __launch_bounds__(256)
void edge_score(const int* __restrict__ src, const int* __restrict__ dst,
                const float* __restrict__ xl, const float* __restrict__ xr,
                const float* __restrict__ att,
                float* __restrict__ score, unsigned* __restrict__ menc)
{
  long long t = (long long)blockIdx.x * blockDim.x + threadIdx.x;
  if (t >= (long long)N_EDGES * HEADS) return;
  int e = (int)(t >> 2), h = (int)(t & 3);
  int s = src[e], d = dst[e];
  const float* pl = xl + (size_t)s * HCDIM + h * CHAN;
  const float* pr = xr + (size_t)d * HCDIM + h * CHAN;
  const float* pa = att + h * CHAN;
  float acc = 0.0f;
  #pragma unroll
  for (int c = 0; c < CHAN; ++c) {
    float v = pl[c] + pr[c];
    v = (v > 0.0f) ? v : v * SLOPE;         // LeakyReLU
    acc = fmaf(pa[c], v, acc);
  }
  score[t] = acc;
  atomicMax(&menc[(size_t)d * HEADS + h], fenc(acc));
}

__global__ __launch_bounds__(256)
void edge_expsum(const int* __restrict__ dst, float* __restrict__ score,
                 const unsigned* __restrict__ menc, float* __restrict__ denom)
{
  long long t = (long long)blockIdx.x * blockDim.x + threadIdx.x;
  if (t >= (long long)N_EDGES * HEADS) return;
  int e = (int)(t >> 2), h = (int)(t & 3);
  int d = dst[e];
  float m  = fdec(menc[(size_t)d * HEADS + h]);
  float ex = __expf(score[t] - m);
  score[t] = ex;
  atomicAdd(&denom[(size_t)d * HEADS + h], ex);
}

__global__ __launch_bounds__(256)
void edge_aggregate(const int* __restrict__ src, const int* __restrict__ dst,
                    const float* __restrict__ score, const float* __restrict__ denom,
                    const float* __restrict__ xl, float* __restrict__ out)
{
  long long t = (long long)blockIdx.x * blockDim.x + threadIdx.x;
  if (t >= (long long)N_EDGES * HEADS) return;
  int e = (int)(t >> 2), h = (int)(t & 3);
  int s = src[e], d = dst[e];
  float alpha = score[t] / (denom[(size_t)d * HEADS + h] + 1e-16f);
  const float* pl = xl  + (size_t)s * HCDIM + h * CHAN;
  float*       po = out + (size_t)d * HCDIM + h * CHAN;
  #pragma unroll
  for (int c = 0; c < CHAN; ++c) atomicAdd(&po[c], pl[c] * alpha);
}

// =====================================================================
// Utility / epilogue kernels
// =====================================================================
__global__ __launch_bounds__(256)
void fill_f32(float* p, float v, long long n) {
  long long t = (long long)blockIdx.x * blockDim.x + threadIdx.x;
  if (t < n) p[t] = v;
}
__global__ __launch_bounds__(256)
void fill_u32(unsigned* p, unsigned v, long long n) {
  long long t = (long long)blockIdx.x * blockDim.x + threadIdx.x;
  if (t < n) p[t] = v;
}
__global__ __launch_bounds__(256)
void init_bias_rows(float* p, const float* __restrict__ bias, long long n) {
  long long t = (long long)blockIdx.x * blockDim.x + threadIdx.x;
  if (t < n) p[t] = bias[t & (HCDIM - 1)];
}
__global__ __launch_bounds__(256)
void relu_inplace(float* p, long long n) {
  long long t = (long long)blockIdx.x * blockDim.x + threadIdx.x;
  if (t < n) p[t] = fmaxf(p[t], 0.0f);
}

// one thread per (node, 32-channel chunk); chunk 0 also counts the node
__global__ __launch_bounds__(256)
void pool_kernel(const int* __restrict__ batch, const float* __restrict__ h,
                 float* __restrict__ pool, float* __restrict__ cnt)
{
  long long t = (long long)blockIdx.x * blockDim.x + threadIdx.x;
  if (t >= (long long)N_NODES * 4) return;
  int n = (int)(t >> 2), ch = (int)(t & 3) * CHAN;
  int g = batch[n];
  const float* ph = h    + (size_t)n * HCDIM + ch;
  float*       pp = pool + (size_t)g * HCDIM + ch;
  #pragma unroll
  for (int c = 0; c < CHAN; ++c) atomicAdd(&pp[c], ph[c]);
  if ((t & 3) == 0) atomicAdd(&cnt[g], 1.0f);
}

__global__ __launch_bounds__(256)
void mean_kernel(const float* __restrict__ pool, const float* __restrict__ cnt,
                 float* __restrict__ g)
{
  long long t = (long long)blockIdx.x * blockDim.x + threadIdx.x;
  if (t >= (long long)N_GRAPH * HCDIM) return;
  int gi = (int)(t >> 7);
  float c = cnt[gi];
  g[t] = pool[t] / fmaxf(c, 1.0f);
}

__global__ __launch_bounds__(256)
void fc3_kernel(const float* __restrict__ g, const float* __restrict__ W,
                const float* __restrict__ b, float* __restrict__ out)
{
  int t = blockIdx.x * blockDim.x + threadIdx.x;
  if (t >= N_GRAPH * 2) return;
  int gi = t >> 1, o = t & 1;
  float acc = b[o];
  const float* pg = g + (size_t)gi * HCDIM;
  #pragma unroll 8
  for (int k = 0; k < HCDIM; ++k) acc = fmaf(pg[k], W[k * 2 + o], acc);
  out[t] = acc;
}

// =====================================================================
extern "C" void kernel_launch(void* const* d_in, const int* in_sizes, int n_in,
                              void* d_out, int out_size, void* d_ws, size_t ws_size,
                              hipStream_t stream)
{
  (void)in_sizes; (void)n_in; (void)out_size; (void)ws_size;

  const float* x       = (const float*)d_in[0];
  const int*   ei      = (const int*)  d_in[1];
  const int*   batch   = (const int*)  d_in[2];
  const float* l1_Wl   = (const float*)d_in[3];
  const float* l1_bl   = (const float*)d_in[4];
  const float* l1_Wr   = (const float*)d_in[5];
  const float* l1_br   = (const float*)d_in[6];
  const float* l1_att  = (const float*)d_in[7];
  const float* l1_bias = (const float*)d_in[8];
  const float* l2_Wl   = (const float*)d_in[9];
  const float* l2_bl   = (const float*)d_in[10];
  const float* l2_Wr   = (const float*)d_in[11];
  const float* l2_br   = (const float*)d_in[12];
  const float* l2_att  = (const float*)d_in[13];
  const float* l2_bias = (const float*)d_in[14];
  const float* fc1_W   = (const float*)d_in[15];
  const float* fc1_b   = (const float*)d_in[16];
  const float* bn1_g   = (const float*)d_in[17];
  const float* bn1_b   = (const float*)d_in[18];
  const float* fc2_W   = (const float*)d_in[19];
  const float* fc2_b   = (const float*)d_in[20];
  const float* bn2_g   = (const float*)d_in[21];
  const float* bn2_b   = (const float*)d_in[22];
  const float* fc3_W   = (const float*)d_in[23];
  const float* fc3_b   = (const float*)d_in[24];

  const int* src = ei;
  const int* dst = ei + N_EDGES;

  // ---- workspace layout (floats) ----
  const size_t NHC = (size_t)N_NODES * HCDIM;   // 6,553,600
  const size_t EH  = (size_t)N_EDGES * HEADS;   // 6,553,600
  const size_t NH  = (size_t)N_NODES * HEADS;   //   204,800
  const size_t GHC = (size_t)N_GRAPH * HCDIM;   //    65,536

  float*    xl   = (float*)d_ws;
  float*    xr   = xl + NHC;
  float*    sc   = xr + NHC;
  unsigned* me   = (unsigned*)(sc + EH);
  float*    dn   = (float*)(me + NH);
  float*    h    = dn + NH;                      // reused for layer-1 and layer-2 output
  float*    pool = h + NHC;
  float*    cnt  = pool + GHC;
  float*    g    = cnt + N_GRAPH;
  float*    t1   = g + GHC;

  dim3 blk(256);
  const int grid_gemm = N_NODES / (16 * MTILES); // 800
  const int grid_mlp  = N_GRAPH / (16 * MTILES); // 8
  const int grid_eh   = (int)(EH / 256);         // 25600
  const int grid_nhc  = (int)(NHC / 256);        // 25600
  const int grid_nh   = (int)(NH / 256);         // 800

  // ================= Layer 1 =================
  gemm_n128_wmma<DIM_IN, 0><<<grid_gemm, blk, 0, stream>>>(x, l1_Wl, l1_bl, nullptr, nullptr, xl);
  gemm_n128_wmma<DIM_IN, 0><<<grid_gemm, blk, 0, stream>>>(x, l1_Wr, l1_br, nullptr, nullptr, xr);
  fill_u32<<<grid_nh, blk, 0, stream>>>(me, ENC_NEG_INF, (long long)NH);
  fill_f32<<<grid_nh, blk, 0, stream>>>(dn, 0.0f, (long long)NH);
  init_bias_rows<<<grid_nhc, blk, 0, stream>>>(h, l1_bias, (long long)NHC);
  edge_score    <<<grid_eh, blk, 0, stream>>>(src, dst, xl, xr, l1_att, sc, me);
  edge_expsum   <<<grid_eh, blk, 0, stream>>>(dst, sc, me, dn);
  edge_aggregate<<<grid_eh, blk, 0, stream>>>(src, dst, sc, dn, xl, h);
  relu_inplace  <<<grid_nhc, blk, 0, stream>>>(h, (long long)NHC);

  // ================= Layer 2 =================
  gemm_n128_wmma<HCDIM, 0><<<grid_gemm, blk, 0, stream>>>(h, l2_Wl, l2_bl, nullptr, nullptr, xl);
  gemm_n128_wmma<HCDIM, 0><<<grid_gemm, blk, 0, stream>>>(h, l2_Wr, l2_br, nullptr, nullptr, xr);
  fill_u32<<<grid_nh, blk, 0, stream>>>(me, ENC_NEG_INF, (long long)NH);
  fill_f32<<<grid_nh, blk, 0, stream>>>(dn, 0.0f, (long long)NH);
  init_bias_rows<<<grid_nhc, blk, 0, stream>>>(h, l2_bias, (long long)NHC);
  edge_score    <<<grid_eh, blk, 0, stream>>>(src, dst, xl, xr, l2_att, sc, me);
  edge_expsum   <<<grid_eh, blk, 0, stream>>>(dst, sc, me, dn);
  edge_aggregate<<<grid_eh, blk, 0, stream>>>(src, dst, sc, dn, xl, h);
  relu_inplace  <<<grid_nhc, blk, 0, stream>>>(h, (long long)NHC);

  // ================= Global mean pool =================
  fill_f32<<<(int)(GHC / 256), blk, 0, stream>>>(pool, 0.0f, (long long)GHC);
  fill_f32<<<(N_GRAPH + 255) / 256, blk, 0, stream>>>(cnt, 0.0f, (long long)N_GRAPH);
  pool_kernel<<<(int)((size_t)N_NODES * 4 / 256), blk, 0, stream>>>(batch, h, pool, cnt);
  mean_kernel<<<(int)(GHC / 256), blk, 0, stream>>>(pool, cnt, g);

  // ================= MLP head =================
  gemm_n128_wmma<HCDIM, 1><<<grid_mlp, blk, 0, stream>>>(g,  fc1_W, fc1_b, bn1_g, bn1_b, t1);
  gemm_n128_wmma<HCDIM, 1><<<grid_mlp, blk, 0, stream>>>(t1, fc2_W, fc2_b, bn2_g, bn2_b, g);
  fc3_kernel<<<(N_GRAPH * 2 + 255) / 256, blk, 0, stream>>>(g, fc3_W, fc3_b, (float*)d_out);
}
